// CachedRoPEAttention_75230647157610
// MI455X (gfx1250) — matrix operand order repaired and checked by reading, not compiled, not run on hardware
//
#include <hip/hip_runtime.h>
#include <hip/hip_bf16.h>

typedef __bf16 bf16_t;
typedef __attribute__((ext_vector_type(16))) __bf16 v16bf;
typedef __attribute__((ext_vector_type(8)))  __bf16 v8bf;
typedef __attribute__((ext_vector_type(4)))  __bf16 v4bf;
typedef __attribute__((ext_vector_type(8)))  float  v8f;

#define D_MODEL 1024
#define N_HEADS 16
#define D_HEAD  64
#define SEQ_T   2048
#define BATCH   2
#define M_TOT   (BATCH * SEQ_T)   // 4096

// ---------------------------------------------------------------------------
// Kernel 0: fp32 -> bf16 conversion (vectorized, n must be a multiple of 4)
// ---------------------------------------------------------------------------
__global__ void cvt_f32_bf16_kernel(const float* __restrict__ src,
                                    bf16_t* __restrict__ dst, int n) {
    int i = (blockIdx.x * blockDim.x + threadIdx.x) * 4;
    if (i + 3 < n) {
        float4 v = *(const float4*)(src + i);
        v4bf o;
        o[0] = (bf16_t)v.x; o[1] = (bf16_t)v.y;
        o[2] = (bf16_t)v.z; o[3] = (bf16_t)v.w;
        *(v4bf*)(dst + i) = o;
    }
}

// ---------------------------------------------------------------------------
// Kernel 1: QKV projection (bf16 WMMA, fp32 accum) + fused RoPE
//   grid = (32, N_HEADS, 3), block = 256 (8 waves); wave -> 16x64 tile
//   mat 0: Q (rope + 1/sqrt(64) scale) -> Qr[b,h,t,dh]
//   mat 1: K (rope)                    -> Kr[b,h,t,dh]
//   mat 2: V                           -> Vt[b,h,dh,t] (transposed)
// ---------------------------------------------------------------------------
__global__ __launch_bounds__(256)
void qkv_rope_kernel(const bf16_t* __restrict__ xb,
                     const bf16_t* __restrict__ wq,
                     const bf16_t* __restrict__ wk,
                     const bf16_t* __restrict__ wv,
                     bf16_t* __restrict__ Qr,
                     bf16_t* __restrict__ Kr,
                     bf16_t* __restrict__ Vt) {
    const int wave = threadIdx.x >> 5;
    const int lane = threadIdx.x & 31;
    const int half = lane >> 4;
    const int l16  = lane & 15;
    const int mtile = blockIdx.x * 8 + wave;   // 0..255
    const int head  = blockIdx.y;              // 0..15
    const int mat   = blockIdx.z;              // 0:Q 1:K 2:V
    const int m0 = mtile * 16;

    const bf16_t* W = (mat == 0) ? wq : (mat == 1) ? wk : wv;

    v8f acc[4] = {};
    const bf16_t* arow = xb + (size_t)(m0 + l16) * D_MODEL;

    for (int k0 = 0; k0 < D_MODEL; k0 += 32) {
        // A fragment: lane row m=l16; k = half*16 + i (contiguous 16)
        v16bf a;
        {
            v8bf r0 = *(const v8bf*)(arow + k0 + half * 16);
            v8bf r1 = *(const v8bf*)(arow + k0 + half * 16 + 8);
            #pragma unroll
            for (int i = 0; i < 8; i++) { a[i] = r0[i]; a[8 + i] = r1[i]; }
        }
        #pragma unroll
        for (int j = 0; j < 4; j++) {
            // B[k,n] = W[n,k] ; lane col n=l16, k = half*16 + i (contiguous 16)
            const bf16_t* brow =
                W + (size_t)(head * D_HEAD + 16 * j + l16) * D_MODEL + k0 + half * 16;
            v8bf c0 = *(const v8bf*)(brow);
            v8bf c1 = *(const v8bf*)(brow + 8);
            v16bf b;
            #pragma unroll
            for (int i = 0; i < 8; i++) { b[i] = c0[i]; b[8 + i] = c1[i]; }
            acc[j] = __builtin_amdgcn_wmma_f32_16x16x32_bf16(
                false, a, false, b, (short)0, acc[j], false, false);
        }
    }

    const int bidx = m0 / SEQ_T;
    const int t0   = m0 % SEQ_T;

    if (mat < 2) {
        // RoPE: element (i,lane) of acc[j] is row t=t0+i+8*half, col e=16*j+l16.
        // Pair (j, j+2) are rotary partners (e and e+32), same lane/element.
        const float oscale = (mat == 0) ? 0.125f : 1.0f;  // fold 1/sqrt(64) into Q
        #pragma unroll
        for (int j = 0; j < 2; j++) {
            const int jj = 16 * j + l16;                       // 0..31
            const float theta = __expf(-(float)jj * 0.28782313662425576f); // ln(1e4)/32
            #pragma unroll
            for (int i = 0; i < 8; i++) {
                float t = (float)(t0 + i + 8 * half);
                float c, s;
                __sincosf(t * theta, &s, &c);  // (sin, cos)
                float v1 = acc[j][i], v2 = acc[j + 2][i];
                acc[j][i]     = (v1 * c - v2 * s) * oscale;
                acc[j + 2][i] = (v2 * c + v1 * s) * oscale;
            }
        }
        bf16_t* dst = (mat == 0) ? Qr : Kr;
        #pragma unroll
        for (int j = 0; j < 4; j++) {
            const int dh = 16 * j + l16;
            bf16_t* p = dst +
                ((size_t)(bidx * N_HEADS + head) * SEQ_T + t0 + 8 * half) * D_HEAD + dh;
            #pragma unroll
            for (int i = 0; i < 8; i++) p[(size_t)i * D_HEAD] = (bf16_t)acc[j][i];
        }
    } else {
        // V transposed: Vt[(bh*64 + dh)*T + t]; t contiguous over i -> packed b128
        #pragma unroll
        for (int j = 0; j < 4; j++) {
            const int dh = 16 * j + l16;
            v8bf pk;
            #pragma unroll
            for (int i = 0; i < 8; i++) pk[i] = (bf16_t)acc[j][i];
            bf16_t* p = Vt +
                ((size_t)(bidx * N_HEADS + head) * D_HEAD + dh) * SEQ_T + t0 + 8 * half;
            *(v8bf*)p = pk;
        }
    }
}

// ---------------------------------------------------------------------------
// Kernel 2: causal flash attention, one 16-row q-tile per wave
//   grid = 512, block = 256 (8 waves). Streams keys in 32-wide chunks.
// ---------------------------------------------------------------------------
__global__ __launch_bounds__(256)
void attn_kernel(const bf16_t* __restrict__ Qr,
                 const bf16_t* __restrict__ Kr,
                 const bf16_t* __restrict__ Vt,
                 bf16_t* __restrict__ Ao) {
    __shared__ __align__(16) bf16_t Plds[8][16 * 32];

    const int wave = threadIdx.x >> 5;
    const int lane = threadIdx.x & 31;
    const int half = lane >> 4;
    const int l16  = lane & 15;
    const int w     = blockIdx.x * 8 + wave;   // 0..4095
    const int qtile = w & 127;
    const int bh    = w >> 7;                  // b*16 + h
    const int q0    = qtile * 16;

    const bf16_t* qbase = Qr + (size_t)bh * SEQ_T * D_HEAD;
    const bf16_t* kbase = Kr + (size_t)bh * SEQ_T * D_HEAD;
    const bf16_t* vbase = Vt + (size_t)bh * D_HEAD * SEQ_T;

    // Q A-fragments, 2 k-steps over D_HEAD=64 (loaded once, pre-scaled)
    v16bf aq[2];
    {
        const bf16_t* qrow = qbase + (size_t)(q0 + l16) * D_HEAD;
        #pragma unroll
        for (int s = 0; s < 2; s++) {
            v8bf r0 = *(const v8bf*)(qrow + 32 * s + half * 16);
            v8bf r1 = *(const v8bf*)(qrow + 32 * s + half * 16 + 8);
            #pragma unroll
            for (int i = 0; i < 8; i++) { aq[s][i] = r0[i]; aq[s][8 + i] = r1[i]; }
        }
    }

    float mrow[8], lrow[8];
    v8f o[4] = {};
    #pragma unroll
    for (int i = 0; i < 8; i++) { mrow[i] = -3.0e38f; lrow[i] = 0.0f; }

    const int nchunk = (q0 + 15) / 32 + 1;     // causal: keys [0, q0+16)
    for (int c = 0; c < nchunk; ++c) {
        const int kt0 = c * 32;
        v8f S[2];
        #pragma unroll
        for (int s = 0; s < 2; s++) {
            const bf16_t* krow =
                kbase + (size_t)(kt0 + 16 * s + l16) * D_HEAD + half * 16;
            v16bf b0, b1;
            {
                v8bf c0 = *(const v8bf*)(krow);
                v8bf c1 = *(const v8bf*)(krow + 8);
                v8bf d0 = *(const v8bf*)(krow + 32);
                v8bf d1 = *(const v8bf*)(krow + 40);
                #pragma unroll
                for (int i = 0; i < 8; i++) {
                    b0[i] = c0[i]; b0[8 + i] = c1[i];
                    b1[i] = d0[i]; b1[8 + i] = d1[i];
                }
            }
            v8f sacc = {};
            sacc = __builtin_amdgcn_wmma_f32_16x16x32_bf16(
                false, aq[0], false, b0, (short)0, sacc, false, false);
            sacc = __builtin_amdgcn_wmma_f32_16x16x32_bf16(
                false, aq[1], false, b1, (short)0, sacc, false, false);
            const int kt = kt0 + 16 * s + l16;
            #pragma unroll
            for (int i = 0; i < 8; i++) {
                int qt = q0 + i + 8 * half;
                if (kt > qt) sacc[i] = -3.0e38f;   // causal mask
            }
            S[s] = sacc;
        }

        // online softmax: rows split per 16-lane half; xor masks < 16 stay in-half
        #pragma unroll
        for (int i = 0; i < 8; i++) {
            float mx = fmaxf(S[0][i], S[1][i]);
            #pragma unroll
            for (int d = 1; d < 16; d <<= 1) mx = fmaxf(mx, __shfl_xor(mx, d, 32));
            float mnew = fmaxf(mrow[i], mx);
            float corr = __expf(mrow[i] - mnew);
            float p0 = __expf(S[0][i] - mnew);
            float p1 = __expf(S[1][i] - mnew);
            S[0][i] = p0; S[1][i] = p1;
            float rs = p0 + p1;
            #pragma unroll
            for (int d = 1; d < 16; d <<= 1) rs += __shfl_xor(rs, d, 32);
            lrow[i] = lrow[i] * corr + rs;
            mrow[i] = mnew;
            o[0][i] *= corr; o[1][i] *= corr; o[2][i] *= corr; o[3][i] *= corr;
        }

        // D-layout -> A-layout reshape of P through wave-private LDS tile
        bf16_t* pw = &Plds[wave][0];
        #pragma unroll
        for (int s = 0; s < 2; s++)
            #pragma unroll
            for (int i = 0; i < 8; i++)
                pw[(i + 8 * half) * 32 + 16 * s + l16] = (bf16_t)S[s][i];

        v16bf ap;
        {
            const bf16_t* pr = pw + l16 * 32 + half * 16;
            v8bf r0 = *(const v8bf*)(pr);
            v8bf r1 = *(const v8bf*)(pr + 8);
            #pragma unroll
            for (int i = 0; i < 8; i++) { ap[i] = r0[i]; ap[8 + i] = r1[i]; }
        }

        // O += P @ V  (K=32 per WMMA, 4 dh-tiles)
        #pragma unroll
        for (int d4 = 0; d4 < 4; d4++) {
            const bf16_t* vrow =
                vbase + (size_t)(16 * d4 + l16) * SEQ_T + kt0 + half * 16;
            v8bf c0 = *(const v8bf*)(vrow);
            v8bf c1 = *(const v8bf*)(vrow + 8);
            v16bf bv;
            #pragma unroll
            for (int i = 0; i < 8; i++) { bv[i] = c0[i]; bv[8 + i] = c1[i]; }
            o[d4] = __builtin_amdgcn_wmma_f32_16x16x32_bf16(
                false, ap, false, bv, (short)0, o[d4], false, false);
        }
    }

    // normalize and store to Ao[b*T + t, h*64 + dh] (bf16)
    const int bidx = bh >> 4, h = bh & 15;
    float inv[8];
    #pragma unroll
    for (int i = 0; i < 8; i++) inv[i] = 1.0f / lrow[i];
    #pragma unroll
    for (int d4 = 0; d4 < 4; d4++) {
        const int e = h * D_HEAD + 16 * d4 + l16;
        bf16_t* p = Ao + ((size_t)bidx * SEQ_T + q0 + 8 * half) * D_MODEL + e;
        #pragma unroll
        for (int i = 0; i < 8; i++)
            p[(size_t)i * D_MODEL] = (bf16_t)(o[d4][i] * inv[i]);
    }
}

// ---------------------------------------------------------------------------
// Kernel 3: output projection out = Ao @ Wo^T  (fp32 stores to d_out)
//   grid = (32, 16), block = 256; wave -> 16x64 tile
// ---------------------------------------------------------------------------
__global__ __launch_bounds__(256)
void out_proj_kernel(const bf16_t* __restrict__ Ao,
                     const bf16_t* __restrict__ wo,
                     float* __restrict__ out) {
    const int wave = threadIdx.x >> 5;
    const int lane = threadIdx.x & 31;
    const int half = lane >> 4;
    const int l16  = lane & 15;
    const int mtile = blockIdx.x * 8 + wave;   // 0..255
    const int ng    = blockIdx.y;              // 0..15
    const int m0 = mtile * 16;

    v8f acc[4] = {};
    const bf16_t* arow = Ao + (size_t)(m0 + l16) * D_MODEL;

    for (int k0 = 0; k0 < D_MODEL; k0 += 32) {
        v16bf a;
        {
            v8bf r0 = *(const v8bf*)(arow + k0 + half * 16);
            v8bf r1 = *(const v8bf*)(arow + k0 + half * 16 + 8);
            #pragma unroll
            for (int i = 0; i < 8; i++) { a[i] = r0[i]; a[8 + i] = r1[i]; }
        }
        #pragma unroll
        for (int j = 0; j < 4; j++) {
            const bf16_t* brow =
                wo + (size_t)(ng * 64 + 16 * j + l16) * D_MODEL + k0 + half * 16;
            v8bf c0 = *(const v8bf*)(brow);
            v8bf c1 = *(const v8bf*)(brow + 8);
            v16bf b;
            #pragma unroll
            for (int i = 0; i < 8; i++) { b[i] = c0[i]; b[8 + i] = c1[i]; }
            acc[j] = __builtin_amdgcn_wmma_f32_16x16x32_bf16(
                false, a, false, b, (short)0, acc[j], false, false);
        }
    }
    #pragma unroll
    for (int j = 0; j < 4; j++) {
        float* p = out + ((size_t)m0 + 8 * half) * D_MODEL + ng * 64 + 16 * j + l16;
        #pragma unroll
        for (int i = 0; i < 8; i++) p[(size_t)i * D_MODEL] = acc[j][i];
    }
}

// ---------------------------------------------------------------------------
extern "C" void kernel_launch(void* const* d_in, const int* in_sizes, int n_in,
                              void* d_out, int out_size, void* d_ws, size_t ws_size,
                              hipStream_t stream) {
    const float* x  = (const float*)d_in[0];
    const float* Wq = (const float*)d_in[1];
    const float* Wk = (const float*)d_in[2];
    const float* Wv = (const float*)d_in[3];
    const float* Wo = (const float*)d_in[4];
    float* out = (float*)d_out;

    const size_t NX = (size_t)M_TOT * D_MODEL;     // 4M elements
    const size_t NW = (size_t)D_MODEL * D_MODEL;   // 1M elements

    bf16_t* xb  = (bf16_t*)d_ws;
    bf16_t* wqb = xb  + NX;
    bf16_t* wkb = wqb + NW;
    bf16_t* wvb = wkb + NW;
    bf16_t* wob = wvb + NW;
    bf16_t* Qr  = wob + NW;   // [B,H,T,64]
    bf16_t* Kr  = Qr  + NX;   // [B,H,T,64]
    bf16_t* Vt  = Kr  + NX;   // [B,H,64,T]
    bf16_t* Ao  = Vt  + NX;   // [B*T, D_MODEL]

    // 0) fp32 -> bf16 staging
    cvt_f32_bf16_kernel<<<(int)(NX / 1024), 256, 0, stream>>>(x,  xb,  (int)NX);
    cvt_f32_bf16_kernel<<<(int)(NW / 1024), 256, 0, stream>>>(Wq, wqb, (int)NW);
    cvt_f32_bf16_kernel<<<(int)(NW / 1024), 256, 0, stream>>>(Wk, wkb, (int)NW);
    cvt_f32_bf16_kernel<<<(int)(NW / 1024), 256, 0, stream>>>(Wv, wvb, (int)NW);
    cvt_f32_bf16_kernel<<<(int)(NW / 1024), 256, 0, stream>>>(Wo, wob, (int)NW);

    // 1) QKV projections + RoPE (Q pre-scaled by 1/sqrt(64))
    qkv_rope_kernel<<<dim3(32, N_HEADS, 3), 256, 0, stream>>>(
        xb, wqb, wkb, wvb, Qr, Kr, Vt);

    // 2) causal flash attention
    attn_kernel<<<512, 256, 0, stream>>>(Qr, Kr, Vt, Ao);

    // 3) output projection
    out_proj_kernel<<<dim3(32, 16), 256, 0, stream>>>(Ao, wob, out);
}
